// GATTemporal_87952340287523
// MI455X (gfx1250) — matrix operand order repaired and checked by reading, not compiled, not run on hardware
//
#include <hip/hip_runtime.h>
#include <hip/hip_bf16.h>
#include <math.h>

// ---------------------------------------------------------------------------
// GAT-temporal pipeline for gfx1250 (MI455X).
//  - Dense GEMMs (x@W1, h@W2) via v_wmma_f32_16x16x32_f16 (f16 in, f32 acc).
//  - B operand staged to LDS with GLOBAL_LOAD_ASYNC_TO_LDS_B128 (ASYNCcnt).
//  - Edge segment-softmax via flat atomic kernels (bandwidth-bound phase),
//    with global_prefetch_b8 on the gathered source rows.
//  - Fused conv1d(3)x2 + BN + avgpool + MLP head, one block per 4 nodes.
// ---------------------------------------------------------------------------

typedef _Float16 half_t;
typedef __attribute__((ext_vector_type(16))) _Float16 v16h;
typedef __attribute__((ext_vector_type(8)))  _Float16 v8h;
typedef __attribute__((ext_vector_type(8)))  float    v8f;

#define NNODES 50000
#define TSTEPS 6
#define NEDGES 800000
#define NHEADS 4

// ---------------- weight cast fp32 -> f16, transposed to [n][k] ----------------
// W layout in memory: [k][n] (k = 0..63). We store wt[n*64 + k] so the GEMM's
// LDS B tile ([n][k]) is a straight contiguous copy -> async-LDS eligible.
__global__ void k_cast_w_gat87952340(const float* __restrict__ W1,
                                     const float* __restrict__ W2,
                                     half_t* __restrict__ w1t,
                                     half_t* __restrict__ w2t) {
  int idx = blockIdx.x * 256 + threadIdx.x;
  if (idx < 64 * 64) {
    int n = idx >> 6, k = idx & 63;
    w1t[idx] = (half_t)W1[k * 64 + n];
  }
  int j = idx - 64 * 64;
  if (j >= 0 && j < 64 * 256) {
    int n = j >> 6, k = j & 63;
    w2t[j] = (half_t)W2[k * 256 + n];
  }
}

// ---------------- generic u32 fill ----------------
__global__ void k_fill_gat87952340(unsigned* __restrict__ p, unsigned v, int count) {
  int idx = blockIdx.x * 256 + threadIdx.x;
  if (idx < count) p[idx] = v;
}

// ---------------- WMMA GEMM: C[Nrows x NOUT] = A[Nrows x 64] * B[64 x NOUT] ---
// A fp32 in global (cast to f16 while staging to LDS); Bt pre-cast f16, [n][k].
// 256 threads = 8 waves; block covers a 32-row stripe and all NOUT columns.
template <int NOUT>
__global__ void k_gemm_wmma_gat87952340(const float* __restrict__ A,
                                        const half_t* __restrict__ Bt,
                                        float* __restrict__ C, int Nrows) {
  __shared__ __attribute__((aligned(32))) half_t sA[32][64];     // rows x K
  __shared__ __attribute__((aligned(32))) half_t sB[NOUT][64];   // n-major: [n][k]
  const int tid = threadIdx.x;
  const int rowBase = blockIdx.x * 32;

  // stage B: straight copy via async LDS DMA, 16B per lane-op (ASYNCcnt)
  {
    const unsigned ldsBase = (unsigned)(uintptr_t)(&sB[0][0]);
    const unsigned long long gBase = (unsigned long long)(uintptr_t)Bt;
    for (int chunk = tid; chunk < NOUT * 8; chunk += 256) {
      unsigned lds = ldsBase + (unsigned)chunk * 16u;
      unsigned long long ga = gBase + (unsigned long long)chunk * 16ull;
      asm volatile("global_load_async_to_lds_b128 %0, %1, off"
                   :: "v"(lds), "v"(ga) : "memory");
    }
  }
  // stage A (fp32 -> f16), zero-pad OOB rows
  for (int idx = tid; idx < 32 * 64; idx += 256) {
    int r = idx >> 6, k = idx & 63;
    int gr = rowBase + r;
    float v = (gr < Nrows) ? A[(size_t)gr * 64 + k] : 0.0f;
    sA[r][k] = (half_t)v;
  }
  asm volatile("s_wait_asynccnt 0" ::: "memory");
  __syncthreads();

  const int wave = tid >> 5;
  const int lane = tid & 31;
  const int hi = lane >> 4;           // 0: lanes 0-15, 1: lanes 16-31
  const int numColTiles = NOUT / 16;
  const bool fullBlock = (rowBase + 32) <= Nrows;

  for (int tile = wave; tile < 2 * numColTiles; tile += 8) {
    const int rowTile = tile / numColTiles;
    const int colTile = tile - rowTile * numColTiles;
    v8f acc = {};
#pragma unroll
    for (int kk = 0; kk < 2; ++kk) {
      // --- A fragment: 16x32 f16, contiguous k = 16*hi + s, same order as B ---
      const int m = lane & 15;
      const int kbA = (hi ? 16 : 0) + 32 * kk;
      v8h lo = *(const v8h*)&sA[rowTile * 16 + m][kbA];        // K = kbA .. kbA+7
      v8h hh = *(const v8h*)&sA[rowTile * 16 + m][kbA + 8];    // K = kbA+8 .. +15
      v16h a;
#pragma unroll
      for (int i = 0; i < 8; ++i) { a[i] = lo[i]; a[8 + i] = hh[i]; }
      // --- B fragment: 32x16 f16; lanes 0-15 K=0..15, lanes 16-31 K=16..31 ---
      const int n = colTile * 16 + (lane & 15);
      const int kbB = (hi ? 16 : 0) + 32 * kk;
      v16h b = *(const v16h*)&sB[n][kbB];
      acc = __builtin_amdgcn_wmma_f32_16x16x32_f16(
          false, a, false, b, (short)0, acc, false, false);
    }
    // --- D store: VGPR j holds row (j + 8*hi), col = lane&15 ---
    const int n = colTile * 16 + (lane & 15);
    const int rbase = rowBase + rowTile * 16 + (hi << 3);
    if (fullBlock) {                      // block-uniform: no EXEC gymnastics
#pragma unroll
      for (int j = 0; j < 8; ++j)
        C[(size_t)(rbase + j) * NOUT + n] = acc[j];
    } else {
#pragma unroll
      for (int j = 0; j < 8; ++j) {
        int row = rbase + j;
        if (row < Nrows) C[(size_t)row * NOUT + n] = acc[j];
      }
    }
  }
}

// ---------------- attention coefficients a_src/a_dst per (node, head) -------
__global__ void k_att_gat87952340(const float* __restrict__ xl,
                                  const float* __restrict__ att_s,
                                  const float* __restrict__ att_d,
                                  float* __restrict__ asrc,
                                  float* __restrict__ adst,
                                  int Nn, int H, int Cc) {
  int idx = blockIdx.x * 256 + threadIdx.x;
  if (idx >= Nn * H) return;
  int n = idx / H, h = idx - n * H;
  const float* row = xl + (size_t)n * H * Cc + (size_t)h * Cc;
  float s = 0.f, d = 0.f;
  for (int c = 0; c < Cc; ++c) {
    float v = row[c];
    s += v * att_s[h * Cc + c];
    d += v * att_d[h * Cc + c];
  }
  asrc[idx] = s;
  adst[idx] = d;
}

// order-preserving float<->uint encoding for atomicMax-based segment max
__device__ __forceinline__ unsigned fenc_gat(float f) {
  unsigned u = __float_as_uint(f);
  return (u & 0x80000000u) ? ~u : (u | 0x80000000u);
}
__device__ __forceinline__ float fdec_gat(unsigned u) {
  return (u & 0x80000000u) ? __uint_as_float(u ^ 0x80000000u)
                           : __uint_as_float(~u);
}

__device__ __forceinline__ void edge_sd_gat(int e, const int* __restrict__ ei,
                                            int E, int& s, int& d) {
  if (e < E) { s = ei[e]; d = ei[E + e]; }
  else       { s = e - E; d = e - E; }      // appended self loops
}

// pass 1: leaky-relu logits + segment max (atomicMax on encoded uint)
__global__ void k_edge_logit_gat87952340(const int* __restrict__ ei,
                                         const float* __restrict__ asrc,
                                         const float* __restrict__ adst,
                                         float* __restrict__ ev,
                                         unsigned* __restrict__ segmax,
                                         int E, int EP, int H) {
  int idx = blockIdx.x * 256 + threadIdx.x;
  if (idx >= EP * H) return;
  int e = idx / H, h = idx - e * H;
  int s, d; edge_sd_gat(e, ei, E, s, d);
  float v = asrc[s * H + h] + adst[d * H + h];
  v = (v > 0.f) ? v : 0.2f * v;             // leaky relu, slope 0.2
  ev[idx] = v;
  atomicMax(&segmax[d * H + h], fenc_gat(v));
}

// pass 2: exp(e - max) + segment sum
__global__ void k_edge_exp_gat87952340(const int* __restrict__ ei,
                                       float* __restrict__ ev,
                                       const unsigned* __restrict__ segmax,
                                       float* __restrict__ segsum,
                                       int E, int EP, int H) {
  int idx = blockIdx.x * 256 + threadIdx.x;
  if (idx >= EP * H) return;
  int e = idx / H, h = idx - e * H;
  int s, d; edge_sd_gat(e, ei, E, s, d);
  float m = fdec_gat(segmax[d * H + h]);
  float ex = expf(ev[idx] - m);
  ev[idx] = ex;
  atomicAdd(&segsum[d * H + h], ex);
}

// pass 3: message aggregation out[dst] += alpha * xl[src]
__global__ void k_edge_msg_gat87952340(const int* __restrict__ ei,
                                       const float* __restrict__ ev,
                                       const float* __restrict__ segsum,
                                       const float* __restrict__ xl,
                                       float* __restrict__ outacc,
                                       int E, int EP, int H, int Cc) {
  int idx = blockIdx.x * 256 + threadIdx.x;
  if (idx >= EP * H) return;
  int e = idx / H, h = idx - e * H;
  int s, d; edge_sd_gat(e, ei, E, s, d);
  const float* xs = xl + (size_t)s * H * Cc + (size_t)h * Cc;
  __builtin_prefetch(xs, 0, 1);             // global_prefetch_b8: hide gather
  float alpha = ev[idx] / (segsum[d * H + h] + 1e-16f);
  float* od = outacc + (size_t)d * H * Cc + (size_t)h * Cc;
  for (int c = 0; c < Cc; ++c) atomicAdd(&od[c], alpha * xs[c]);
}

// layer1 epilogue: +bias, ELU (concat heads)
__global__ void k_final_concat_gat87952340(const float* __restrict__ outacc,
                                           const float* __restrict__ bias,
                                           float* __restrict__ out, int count) {
  int idx = blockIdx.x * 256 + threadIdx.x;
  if (idx >= count) return;
  float v = outacc[idx] + bias[idx & 63];
  out[idx] = (v > 0.f) ? v : (expf(v) - 1.f);
}

// layer2 epilogue: mean over heads, +bias, ELU
__global__ void k_final_mean_gat87952340(const float* __restrict__ outacc,
                                         const float* __restrict__ bias,
                                         float* __restrict__ out, int Nn) {
  int idx = blockIdx.x * 256 + threadIdx.x;
  if (idx >= Nn * 64) return;
  int n = idx >> 6, c = idx & 63;
  const float* r = outacc + (size_t)n * 256;
  float v = 0.25f * (r[c] + r[64 + c] + r[128 + c] + r[192 + c]) + bias[c];
  out[idx] = (v > 0.f) ? v : (expf(v) - 1.f);
}

// -------- fused head: conv1d(3,p1)+BN+relu x2, avgpool over T, MLP 64->32->1 -
// 256 threads = 4 nodes x 64 channels.
__global__ void k_head_gat87952340(const float* __restrict__ ht,
                                   const float* __restrict__ cw1, const float* __restrict__ cb1,
                                   const float* __restrict__ g1,  const float* __restrict__ be1,
                                   const float* __restrict__ cw2, const float* __restrict__ cb2,
                                   const float* __restrict__ g2,  const float* __restrict__ be2,
                                   const float* __restrict__ lw1, const float* __restrict__ lb1,
                                   const float* __restrict__ lw2, const float* __restrict__ lb2,
                                   float* __restrict__ out, int Nn) {
  __shared__ float sIn[4][64][8];    // zero-padded T axis (slots 1..6 live)
  __shared__ float sMid[4][64][8];
  __shared__ float sPool[4][64];
  __shared__ float sH[4][32];
  const int tid = threadIdx.x;
  const int node = tid >> 6;
  const int c = tid & 63;
  const int gn = blockIdx.x * 4 + node;
  const bool valid = gn < Nn;
  const float bns = 1.0f / sqrtf(1.0f + 1e-5f);

  sIn[node][c][0] = 0.f; sIn[node][c][7] = 0.f;
  sMid[node][c][0] = 0.f; sMid[node][c][7] = 0.f;
  for (int t = 0; t < TSTEPS; ++t)
    sIn[node][c][t + 1] = valid ? ht[(size_t)t * Nn * 64 + (size_t)gn * 64 + c] : 0.f;
  __syncthreads();

  // conv1 + BN + relu (out channel = c)
  {
    float acc[TSTEPS];
#pragma unroll
    for (int t = 0; t < TSTEPS; ++t) acc[t] = cb1[c];
    for (int i = 0; i < 64; ++i) {
      const float w0 = cw1[(c * 64 + i) * 3 + 0];
      const float w1 = cw1[(c * 64 + i) * 3 + 1];
      const float w2 = cw1[(c * 64 + i) * 3 + 2];
      const float* xi = sIn[node][i];
#pragma unroll
      for (int t = 0; t < TSTEPS; ++t)
        acc[t] += w0 * xi[t] + w1 * xi[t + 1] + w2 * xi[t + 2];
    }
    const float sc = g1[c] * bns, sh = be1[c];
#pragma unroll
    for (int t = 0; t < TSTEPS; ++t) {
      float y = acc[t] * sc + sh;
      sMid[node][c][t + 1] = (y > 0.f) ? y : 0.f;
    }
  }
  __syncthreads();

  // conv2 + BN + relu + mean pool over T
  {
    float acc[TSTEPS];
#pragma unroll
    for (int t = 0; t < TSTEPS; ++t) acc[t] = cb2[c];
    for (int i = 0; i < 64; ++i) {
      const float w0 = cw2[(c * 64 + i) * 3 + 0];
      const float w1 = cw2[(c * 64 + i) * 3 + 1];
      const float w2 = cw2[(c * 64 + i) * 3 + 2];
      const float* xi = sMid[node][i];
#pragma unroll
      for (int t = 0; t < TSTEPS; ++t)
        acc[t] += w0 * xi[t] + w1 * xi[t + 1] + w2 * xi[t + 2];
    }
    const float sc = g2[c] * bns, sh = be2[c];
    float pooled = 0.f;
#pragma unroll
    for (int t = 0; t < TSTEPS; ++t) {
      float y = acc[t] * sc + sh;
      pooled += (y > 0.f) ? y : 0.f;
    }
    sPool[node][c] = pooled * (1.0f / (float)TSTEPS);
  }
  __syncthreads();

  if (c < 32) {
    float hsum = lb1[c];
    for (int o = 0; o < 64; ++o) hsum += sPool[node][o] * lw1[o * 32 + c];
    sH[node][c] = (hsum > 0.f) ? hsum : 0.f;
  }
  __syncthreads();

  if (c == 0 && valid) {
    float logit = lb2[0];
    for (int j = 0; j < 32; ++j) logit += sH[node][j] * lw2[j];
    out[gn] = logit;
  }
}

// ---------------------------------------------------------------------------
extern "C" void kernel_launch(void* const* d_in, const int* in_sizes, int n_in,
                              void* d_out, int out_size, void* d_ws, size_t ws_size,
                              hipStream_t stream) {
  (void)in_sizes; (void)n_in; (void)out_size; (void)ws_size;
  const float* x_seq = (const float*)d_in[0];
  const int*   ei    = (const int*)d_in[1];
  const float* W1  = (const float*)d_in[2];
  const float* as1 = (const float*)d_in[3];
  const float* ad1 = (const float*)d_in[4];
  const float* b1  = (const float*)d_in[5];
  const float* W2  = (const float*)d_in[6];
  const float* as2 = (const float*)d_in[7];
  const float* ad2 = (const float*)d_in[8];
  const float* b2  = (const float*)d_in[9];
  const float* cw1 = (const float*)d_in[10];
  const float* cb1 = (const float*)d_in[11];
  const float* g1  = (const float*)d_in[12];
  const float* be1 = (const float*)d_in[13];
  const float* cw2 = (const float*)d_in[14];
  const float* cb2 = (const float*)d_in[15];
  const float* g2  = (const float*)d_in[16];
  const float* be2 = (const float*)d_in[17];
  const float* lw1 = (const float*)d_in[18];
  const float* lb1 = (const float*)d_in[19];
  const float* lw2 = (const float*)d_in[20];
  const float* lb2 = (const float*)d_in[21];

  const int Nn = NNODES, Ee = NEDGES, H = NHEADS;
  const int EP = Ee + Nn;

  // carve workspace
  char* ws = (char*)d_ws;
  size_t off = 0;
  auto take = [&](size_t bytes) -> void* {
    off = (off + 255) & ~(size_t)255;
    void* p = ws + off;
    off += bytes;
    return p;
  };
  half_t*   w1t    = (half_t*)take((size_t)64 * 64 * sizeof(half_t));
  half_t*   w2t    = (half_t*)take((size_t)64 * 256 * sizeof(half_t));
  float*    xl     = (float*)take((size_t)Nn * 256 * sizeof(float));
  float*    h1     = (float*)take((size_t)Nn * 64 * sizeof(float));
  float*    ht     = (float*)take((size_t)TSTEPS * Nn * 64 * sizeof(float));
  float*    asrc   = (float*)take((size_t)Nn * H * sizeof(float));
  float*    adst   = (float*)take((size_t)Nn * H * sizeof(float));
  unsigned* segmax = (unsigned*)take((size_t)Nn * H * sizeof(unsigned));
  float*    segsum = (float*)take((size_t)Nn * H * sizeof(float));
  float*    ev     = (float*)take((size_t)EP * H * sizeof(float));
  float*    outacc = (float*)take((size_t)Nn * 256 * sizeof(float));

  const int grEH   = (EP * H + 255) / 256;
  const int grNH   = (Nn * H + 255) / 256;
  const int grGemm = (Nn + 31) / 32;

  k_cast_w_gat87952340<<<(64 * 64 + 64 * 256 + 255) / 256, 256, 0, stream>>>(W1, W2, w1t, w2t);

  for (int t = 0; t < TSTEPS; ++t) {
    const float* xt = x_seq + (size_t)t * Nn * 64;

    // ---------------- layer 1 (64 -> 4x16, concat) ----------------
    k_gemm_wmma_gat87952340<64><<<grGemm, 256, 0, stream>>>(xt, w1t, xl, Nn);
    k_att_gat87952340<<<grNH, 256, 0, stream>>>(xl, as1, ad1, asrc, adst, Nn, H, 16);
    k_fill_gat87952340<<<grNH, 256, 0, stream>>>(segmax, 0u, Nn * H);
    k_fill_gat87952340<<<grNH, 256, 0, stream>>>((unsigned*)segsum, 0u, Nn * H);
    k_fill_gat87952340<<<(Nn * 64 + 255) / 256, 256, 0, stream>>>((unsigned*)outacc, 0u, Nn * 64);
    k_edge_logit_gat87952340<<<grEH, 256, 0, stream>>>(ei, asrc, adst, ev, segmax, Ee, EP, H);
    k_edge_exp_gat87952340<<<grEH, 256, 0, stream>>>(ei, ev, segmax, segsum, Ee, EP, H);
    k_edge_msg_gat87952340<<<grEH, 256, 0, stream>>>(ei, ev, segsum, xl, outacc, Ee, EP, H, 16);
    k_final_concat_gat87952340<<<(Nn * 64 + 255) / 256, 256, 0, stream>>>(outacc, b1, h1, Nn * 64);

    // ---------------- layer 2 (64 -> 4x64, mean) ----------------
    k_gemm_wmma_gat87952340<256><<<grGemm, 256, 0, stream>>>(h1, w2t, xl, Nn);
    k_att_gat87952340<<<grNH, 256, 0, stream>>>(xl, as2, ad2, asrc, adst, Nn, H, 64);
    k_fill_gat87952340<<<grNH, 256, 0, stream>>>(segmax, 0u, Nn * H);
    k_fill_gat87952340<<<grNH, 256, 0, stream>>>((unsigned*)segsum, 0u, Nn * H);
    k_fill_gat87952340<<<(Nn * 256 + 255) / 256, 256, 0, stream>>>((unsigned*)outacc, 0u, Nn * 256);
    k_edge_logit_gat87952340<<<grEH, 256, 0, stream>>>(ei, asrc, adst, ev, segmax, Ee, EP, H);
    k_edge_exp_gat87952340<<<grEH, 256, 0, stream>>>(ei, ev, segmax, segsum, Ee, EP, H);
    k_edge_msg_gat87952340<<<grEH, 256, 0, stream>>>(ei, ev, segsum, xl, outacc, Ee, EP, H, 64);
    k_final_mean_gat87952340<<<(Nn * 64 + 255) / 256, 256, 0, stream>>>(
        outacc, b2, ht + (size_t)t * Nn * 64, Nn);
  }

  // ---------------- temporal conv head ----------------
  k_head_gat87952340<<<(Nn + 3) / 4, 256, 0, stream>>>(
      ht, cw1, cb1, g1, be1, cw2, cb2, g2, be2, lw1, lb1, lw2, lb2,
      (float*)d_out, Nn);
}
